// FullAttention_34385508172158
// MI455X (gfx1250) — matrix operand order repaired and checked by reading, not compiled, not run on hardware
//
#include <hip/hip_runtime.h>
#include <hip/hip_bf16.h>

typedef __attribute__((ext_vector_type(16))) _Float16 v16h;
typedef __attribute__((ext_vector_type(8)))  _Float16 v8h;
typedef __attribute__((ext_vector_type(8)))  float    v8f;
typedef __attribute__((ext_vector_type(4)))  float    v4f;
typedef __attribute__((ext_vector_type(4)))  unsigned v4u;

#define B_ 2
#define L_ 2048
#define H_ 16
#define E_ 64
#define RS_ (H_ * E_)     // floats between consecutive sequence positions

#define SK_STRIDE  72     // padded LDS row stride (halves) for K  [32 x 64]
#define SVT_STRIDE 40     // padded LDS row stride (halves) for V^T [64 x 32]

#if __has_builtin(__builtin_amdgcn_exp2f)
#define EXP2F(x) __builtin_amdgcn_exp2f(x)
#else
#define EXP2F(x) exp2f(x)
#endif

// softmax scale folded into log2 domain: 1/sqrt(64) * log2(e)
#define KSCALE 0.18033688011112042f

// ds_swizzle SWAPX16: group-of-32 mode, xor_mask=0x10, or=0, and_mask=0x1f
#define SWZ_SWAPX16 0x401f

__device__ __forceinline__ float xor16f(float x) {
    return __int_as_float(__builtin_amdgcn_ds_swizzle(__float_as_int(x), SWZ_SWAPX16));
}
__device__ __forceinline__ unsigned xor16u(unsigned x) {
    return (unsigned)__builtin_amdgcn_ds_swizzle((int)x, SWZ_SWAPX16);
}

union U8H { v8h h; v4u u; };
union U16H { v16h h; v4u u[2]; };

// One 32-key step for one wave's 16-query tile. MASKED=false is the steady-state
// fast path; MASKED=true handles the 1-2 diagonal-boundary blocks. Callers use
// wave-uniform (scalar) branches, so EXEC is all-1s at every WMMA.
template <bool MASKED>
__device__ __forceinline__ void tile_step(int jb, int n, int hi, int q_glob,
                                          const _Float16* __restrict__ sK,
                                          const _Float16* __restrict__ sVT,
                                          const v16h* qb, v8f* acc,
                                          float& m_run, float& l_run)
{
    // ---- K A-tiles from LDS: A[key][e], half t -> e = c*32 + hi*16 + t
    v16h ka[2][2];
#pragma unroll
    for (int s = 0; s < 2; ++s) {
        const _Float16* kr = &sK[(s * 16 + n) * SK_STRIDE];
#pragma unroll
        for (int c = 0; c < 2; ++c) {
            v8h lo  = *(const v8h*)(kr + c * 32 + hi * 16);
            v8h hi8 = *(const v8h*)(kr + c * 32 + hi * 16 + 8);
            ka[s][c] = __builtin_shufflevector(lo, hi8, 0,1,2,3,4,5,6,7,8,9,10,11,12,13,14,15);
        }
    }

    // ---- S^T = K x Q^T ----
    v8f c0 = {}, c1 = {};
    c0 = __builtin_amdgcn_wmma_f32_16x16x32_f16(false, ka[0][0], false, qb[0], (short)0, c0, false, false);
    c0 = __builtin_amdgcn_wmma_f32_16x16x32_f16(false, ka[0][1], false, qb[1], (short)0, c0, false, false);
    c1 = __builtin_amdgcn_wmma_f32_16x16x32_f16(false, ka[1][0], false, qb[0], (short)0, c1, false, false);
    c1 = __builtin_amdgcn_wmma_f32_16x16x32_f16(false, ka[1][1], false, qb[1], (short)0, c1, false, false);

    // ---- scale (+ causal mask on boundary tiles); online softmax in exp2 domain ----
    float p0[8], p1[8];
    float tmax = -1e30f;
#pragma unroll
    for (int r = 0; r < 8; ++r) {
        float s0 = c0[r] * KSCALE;
        float s1 = c1[r] * KSCALE;
        if (MASKED) {
            const int k0 = jb + r + hi * 8;
            const int k1 = k0 + 16;
            if (k0 > q_glob) s0 = -1e30f;
            if (k1 > q_glob) s1 = -1e30f;
        }
        p0[r] = s0; p1[r] = s1;
        tmax = fmaxf(tmax, fmaxf(s0, s1));
    }
    tmax = fmaxf(tmax, xor16f(tmax));
    const float new_m = fmaxf(m_run, tmax);
    float rsum = 0.0f;
#pragma unroll
    for (int r = 0; r < 8; ++r) {
        p0[r] = EXP2F(p0[r] - new_m);
        p1[r] = EXP2F(p1[r] - new_m);
        rsum += p0[r] + p1[r];
    }
    rsum += xor16f(rsum);
    const float alpha = EXP2F(m_run - new_m);
    l_run = l_run * alpha + rsum;
    m_run = new_m;

    // ---- P^T as WMMA B-tile: half t -> key s = hi*16 + t ----
    // Convert to packed f16 first, then exchange 8 dwords across the xor16 halves
    // and assemble with 8 integer selects.
    U8H ph0, ph1;
#pragma unroll
    for (int r = 0; r < 8; ++r) { ph0.h[r] = (_Float16)p0[r]; ph1.h[r] = (_Float16)p1[r]; }
    v4u s0u, s1u, lo4, hi4;
#pragma unroll
    for (int i = 0; i < 4; ++i) {
        s0u[i] = xor16u(ph0.u[i]);      // partner's c0 pairs
        s1u[i] = xor16u(ph1.u[i]);      // partner's c1 pairs
    }
#pragma unroll
    for (int i = 0; i < 4; ++i) {
        lo4[i] = hi ? s1u[i] : ph0.u[i];   // pb halves t=0..7  (keys jb+hi*16+t)
        hi4[i] = hi ? ph1.u[i] : s0u[i];   // pb halves t=8..15
    }
    U16H pbu; pbu.u[0] = lo4; pbu.u[1] = hi4;
    const v16h pb = pbu.h;

    // ---- O^T += V^T x P^T over 4 d-chunks of 16 ----
#pragma unroll
    for (int dt = 0; dt < 4; ++dt) {
        const _Float16* vr = &sVT[(dt * 16 + n) * SVT_STRIDE];
        v8h lo  = *(const v8h*)(vr + hi * 16);
        v8h hi8 = *(const v8h*)(vr + hi * 16 + 8);
        v16h va = __builtin_shufflevector(lo, hi8, 0,1,2,3,4,5,6,7,8,9,10,11,12,13,14,15);
        v8f a = acc[dt];
#pragma unroll
        for (int r = 0; r < 8; ++r) a[r] *= alpha;   // running-max rescale
        acc[dt] = __builtin_amdgcn_wmma_f32_16x16x32_f16(false, va, false, pb, (short)0, a, false, false);
    }
}

// Flash attention forward, causal. 4 wave32s per block, one 16-query tile each,
// all for the same (b,h). K/V tiles (32 keys) staged cooperatively into LDS as
// f16 (converted once), WMMA operands fed from LDS via ds_load_b128.
__global__ __launch_bounds__(128)
void FullAttention_34385508172158_kernel(const float* __restrict__ Q,
                                         const float* __restrict__ K,
                                         const float* __restrict__ V,
                                         float* __restrict__ O)
{
    __shared__ __align__(32) _Float16 sK [32 * SK_STRIDE];   // K[key][e]
    __shared__ __align__(32) _Float16 sVT[64 * SVT_STRIDE];  // V^T[d][s]

    const int tid  = threadIdx.x;
    const int lane = tid & 31;
    const int wave = tid >> 5;
    const int n    = lane & 15;   // WMMA N index (query col / d row)
    const int hi   = lane >> 4;   // 16-lane half

    const int qblocks = L_ / 64;
    const int bh   = blockIdx.x / qblocks;
    const int qblk = blockIdx.x % qblocks;
    const int b = bh / H_, h = bh % H_;
    // readfirstlane: tell the compiler m0 is wave-uniform so the causal
    // branches become scalar (s_cmp/s_cbranch) -> EXEC provably all-1s at WMMAs
    const int m0 = __builtin_amdgcn_readfirstlane((qblk * 4 + wave) * 16);

    const size_t base = ((size_t)b * L_ * H_ + h) * E_;
    const float* Qb = Q + base;
    const float* Kb = K + base;
    const float* Vb = V + base;
    float*       Ob = O + base;

    // staging work split: K by (row, 16-col chunk), V^T by (d, 16-key chunk)
    const int krow_s = tid >> 2;                  // 0..31
    const int kch    = tid & 3;                   // 0..3
    const int vd     = tid & 63;                  // 0..63
    const int vsh    = tid >> 6;                  // 0..1

    // ---- Q^T resident as two WMMA B-tiles (e-chunks 0-31, 32-63) ----
    // B layout: lane holds column q=n; half t -> e = ebase + hi*16 + t
    const float* qrow = Qb + (size_t)(m0 + n) * RS_;
    v16h qb[2];
#pragma unroll
    for (int c = 0; c < 2; ++c) {
        const v4f* p = (const v4f*)(qrow + c * 32 + hi * 16);
        v4f f0 = p[0], f1 = p[1], f2 = p[2], f3 = p[3];
#pragma unroll
        for (int t = 0; t < 4; ++t) {
            qb[c][t]      = (_Float16)f0[t];
            qb[c][4 + t]  = (_Float16)f1[t];
            qb[c][8 + t]  = (_Float16)f2[t];
            qb[c][12 + t] = (_Float16)f3[t];
        }
    }

    v8f acc[4] = {};                 // O^T tiles: d = dt*16 + r + hi*8, q = n
    float m_run = -1e30f, l_run = 0.0f;

    const int q_glob     = m0 + n;
    const int my_jend    = m0 + 16;          // this wave's causal key limit
    const int block_jend = qblk * 64 + 64;   // block-uniform key limit

    for (int jb = 0; jb < block_jend; jb += 32) {
        __syncthreads();   // previous tile fully consumed before overwrite

        // ---- stage K tile [32 x 64] as f16, row-major ----
        {
            const v4f* kv = (const v4f*)(Kb + (size_t)(jb + krow_s) * RS_ + kch * 16);
            v4f f0 = kv[0], f1 = kv[1], f2 = kv[2], f3 = kv[3];
            v16h kh;
#pragma unroll
            for (int t = 0; t < 4; ++t) {
                kh[t]      = (_Float16)f0[t];
                kh[4 + t]  = (_Float16)f1[t];
                kh[8 + t]  = (_Float16)f2[t];
                kh[12 + t] = (_Float16)f3[t];
            }
            *(v16h*)&sK[krow_s * SK_STRIDE + kch * 16] = kh;
        }
        // ---- stage V^T tile [64 x 32] as f16 (transpose during staging) ----
        {
            const float* vp = Vb + (size_t)(jb + vsh * 16) * RS_ + vd;
            v16h vh;
#pragma unroll
            for (int t = 0; t < 16; ++t)
                vh[t] = (_Float16)vp[(size_t)t * RS_];   // coalesced across lanes (d)
            *(v16h*)&sVT[vd * SVT_STRIDE + vsh * 16] = vh;
        }
        if (jb + 32 < block_jend)   // speculative prefetch of next K tile
            __builtin_prefetch(Kb + (size_t)(jb + 32 + krow_s) * RS_ + kch * 16, 0, 0);

        __syncthreads();

        if (jb < my_jend) {          // scalar, wave-uniform causal skip
            if (jb + 31 <= m0)       // tile fully below diagonal: no masking needed
                tile_step<false>(jb, n, hi, q_glob, sK, sVT, qb, acc, m_run, l_run);
            else                     // diagonal boundary tile
                tile_step<true >(jb, n, hi, q_glob, sK, sVT, qb, acc, m_run, l_run);
        }
    }

    // ---- normalize and write: O[m0+q][d] = acc^T / l (8 contiguous floats/lane/chunk)
    const float inv_l = 1.0f / l_run;
    float* orow = Ob + (size_t)(m0 + n) * RS_;
#pragma unroll
    for (int dt = 0; dt < 4; ++dt) {
        v4f lo  = { acc[dt][0] * inv_l, acc[dt][1] * inv_l, acc[dt][2] * inv_l, acc[dt][3] * inv_l };
        v4f hi4 = { acc[dt][4] * inv_l, acc[dt][5] * inv_l, acc[dt][6] * inv_l, acc[dt][7] * inv_l };
        *(v4f*)(orow + dt * 16 + hi * 8)     = lo;
        *(v4f*)(orow + dt * 16 + hi * 8 + 4) = hi4;
    }
}

extern "C" void kernel_launch(void* const* d_in, const int* in_sizes, int n_in,
                              void* d_out, int out_size, void* d_ws, size_t ws_size,
                              hipStream_t stream) {
    const float* Q = (const float*)d_in[0];
    const float* K = (const float*)d_in[1];
    const float* V = (const float*)d_in[2];
    float*       O = (float*)d_out;
    dim3 grid(B_ * H_ * (L_ / 64));   // 1024 blocks
    dim3 block(128);                  // 4 wave32s
    hipLaunchKernelGGL(FullAttention_34385508172158_kernel, grid, block, 0, stream,
                       Q, K, V, O);
}
